// BilateralAugmentation_30245159698514
// MI455X (gfx1250) — matrix operand order repaired and checked by reading, not compiled, not run on hardware
//
#include <hip/hip_runtime.h>
#include <hip/hip_bf16.h>

typedef __bf16 bf16;
typedef __attribute__((ext_vector_type(16))) __bf16 v16bf;
typedef __attribute__((ext_vector_type(8)))  __bf16 v8bf;
typedef __attribute__((ext_vector_type(8)))  float  v8f;

#define NB    2
#define NPTS  8192
#define KNN   16
#define WAVES 4

// ---------- helpers ----------
__device__ __forceinline__ v8f wmma_bf16(v16bf a, v16bf b, v8f c) {
  // D = A(16x32 bf16) * B(32x16 bf16) + C(16x16 f32)
  return __builtin_amdgcn_wmma_f32_16x16x32_bf16(false, a, false, b, (short)0, c, false, false);
}

__device__ __forceinline__ v8f splat8(float x) {
  v8f r;
#pragma unroll
  for (int i = 0; i < 8; i++) r[i] = x;
  return r;
}

// A fragment (16x32, M=row striped over lanes%16): lane holds
//   elems 0..7  = channels k0 + 8*h + j
//   elems 8..15 = channels k0 + 16 + 8*h + j
__device__ __forceinline__ v16bf load_a_pair(const bf16* row, int k0, int h) {
  v8bf lo = *(const v8bf*)(row + k0 + 8 * h);
  v8bf hi = *(const v8bf*)(row + k0 + 16 + 8 * h);
  return __builtin_shufflevector(lo, hi, 0,1,2,3,4,5,6,7,8,9,10,11,12,13,14,15);
}

// B fragment (32x16, N=col striped over lanes%16): lane holds
//   elems 0..7  = W[col][k0 + 8*h + j]
//   elems 8..15 = W[col][k0 + 16 + 8*h + j]
__device__ __forceinline__ v16bf load_b16(const bf16* W, int stride, int col, int k0, int h) {
  const bf16* p = W + (size_t)col * stride + k0;
  v8bf lo = *(const v8bf*)(p + 8 * h);
  v8bf hi = *(const v8bf*)(p + 16 + 8 * h);
  return __builtin_shufflevector(lo, hi, 0,1,2,3,4,5,6,7,8,9,10,11,12,13,14,15);
}

// per-lane prefetch -> global_prefetch_b8 (warms WGP$/L2 ahead of the WMMA loops)
__device__ __forceinline__ void pf(const void* p) { __builtin_prefetch(p, 0, 2); }

// order LDS stores before subsequent cross-lane LDS loads within a wave
__device__ __forceinline__ void wave_lds_fence() {
  __builtin_amdgcn_wave_barrier();
  asm volatile("s_wait_dscnt 0" ::: "memory");
  __builtin_amdgcn_wave_barrier();
}

// ---------- kernel 0: fold BN gamma into weights, convert fp32 -> bf16 ----------
__global__ void prep_weights(const float* w1, const float* g1,
                             const float* w5, const float* g5,
                             const float* w6, const float* g6,
                             const float* w7, const float* g7,
                             const float* w8, const float* g8,
                             const float* w9,
                             const float* w10, const float* g10,
                             const float* w11, const float* g11,
                             bf16* W1, bf16* W5, bf16* W6, bf16* W7,
                             bf16* W8, bf16* W9, bf16* W10, bf16* W11) {
  int tid = blockIdx.x * blockDim.x + threadIdx.x;
  int nth = gridDim.x * blockDim.x;
  for (int i = tid; i < 64 * 64; i += nth) W1[i] = (bf16)(g1[i / 64] * w1[i]);
  for (int i = tid; i < 16 * 128; i += nth) {
    int o = i / 128, c = i % 128;
    W5[i] = (bf16)((o < 3) ? g5[o] * w5[o * 128 + c] : 0.f);
  }
  for (int i = tid; i < 64 * 32; i += nth) {
    int o = i / 32, c = i % 32;
    W6[i] = (bf16)((c < 9) ? g6[o] * w6[o * 9 + c] : 0.f);
    W7[i] = (bf16)((c < 9) ? g7[o] * w7[o * 9 + c] : 0.f);
  }
  for (int i = tid; i < 64 * 192; i += nth) W8[i] = (bf16)(g8[i / 192] * w8[i]);
  for (int i = tid; i < 128 * 128; i += nth) W9[i] = (bf16)w9[i];
  for (int i = tid; i < 128 * 256; i += nth) W10[i] = (bf16)(g10[i / 256] * w10[i]);
  for (int i = tid; i < 256 * 128; i += nth) W11[i] = (bf16)(g11[i / 128] * w11[i]);
}

// ---------- kernel 1: f3 = relu(W1' @ feature + be1), stored point-major bf16 ----------
__global__ void k1_feat(const float* feature, const bf16* W1, const float* be1, bf16* f3bf) {
  __shared__ bf16 tr[WAVES][16][72];
  int wave = threadIdx.x >> 5, lane = threadIdx.x & 31;
  int nn = lane & 15, h = lane >> 4;
  int tile = blockIdx.x * WAVES + wave;          // 0..1023
  int b = tile / (NPTS / 16);
  int n0 = (tile % (NPTS / 16)) * 16;

  // warm weight lines while the strided feature gather is in flight
#pragma unroll
  for (int t = 0; t < 4; t++) pf(W1 + (size_t)(16 * t + nn) * 64);

  v8f acc[4];
#pragma unroll
  for (int t = 0; t < 4; t++) acc[t] = splat8(be1[16 * t + nn]);

#pragma unroll
  for (int k0 = 0; k0 < 64; k0 += 32) {
    v16bf a;
#pragma unroll
    for (int j = 0; j < 8; j++) {
      int c0 = k0 + 8 * h + j;
      int c1 = k0 + 16 + 8 * h + j;
      a[j]     = (bf16)feature[((size_t)b * 64 + c0) * NPTS + n0 + nn];
      a[8 + j] = (bf16)feature[((size_t)b * 64 + c1) * NPTS + n0 + nn];
    }
#pragma unroll
    for (int t = 0; t < 4; t++) {
      v16bf bb = load_b16(W1, 64, 16 * t + nn, k0, h);
      acc[t] = wmma_bf16(a, bb, acc[t]);
    }
  }
#pragma unroll
  for (int t = 0; t < 4; t++)
#pragma unroll
    for (int v = 0; v < 8; v++) {
      float x = acc[t][v];
      tr[wave][v + 8 * h][16 * t + nn] = (bf16)(x > 0.f ? x : 0.f);
    }
  wave_lds_fence();
  const bf16* src = &tr[wave][nn][32 * h];
  bf16* dst = f3bf + ((size_t)b * NPTS + n0 + nn) * 64 + 32 * h;
#pragma unroll
  for (int q = 0; q < 4; q++) *(v8bf*)(dst + 8 * q) = *(const v8bf*)(src + 8 * q);
}

// ---------- kernel 2: fused per-point neighbor pipeline ----------
__global__ void k2_point(const bf16* f3bf, const float* xyz, const int* nidx,
                         const bf16* W5, const float* be5,
                         const bf16* W6, const float* be6,
                         const bf16* W7, const float* be7,
                         const bf16* W8, const float* be8,
                         const bf16* W9, const float* b9,
                         bf16* encbf) {
  __shared__ bf16 st[WAVES][16][200];     // neighbor-major activation staging
  __shared__ bf16 st2[WAVES][16][32];     // xyz_info (9ch, zero-padded to 32)
  __shared__ float xoff[WAVES][16][4];    // per-neighbor xyz offsets
  __shared__ bf16 erow[WAVES][256];       // per-point encoding

  int wave = threadIdx.x >> 5, lane = threadIdx.x & 31;
  int nn = lane & 15, h = lane >> 4;
  int pt = blockIdx.x * WAVES + wave;     // 0..16383
  int b = pt / NPTS, n = pt % NPTS;

  bf16 (*S)[200] = st[wave];

  // warm all weight-fragment rows this lane will consume, overlapping the gather
  pf(W5 + (size_t)nn * 128);
  pf(W6 + (size_t)nn * 32);
  pf(W7 + (size_t)nn * 32);
#pragma unroll
  for (int t = 0; t < 4; t++) {
    pf(W8 + (size_t)(16 * t + nn) * 192);
    pf(W8 + (size_t)(16 * t + nn) * 192 + 128);
  }
#pragma unroll
  for (int t = 0; t < 8; t++) pf(W9 + (size_t)(16 * t + nn) * 128);

  // ---- gather neighbor & center features (bf16) ----
  int idxm = nidx[((size_t)b * NPTS + n) * KNN + nn];
  const bf16* nrow = f3bf + ((size_t)b * NPTS + idxm) * 64 + 32 * h;
  const bf16* trow = f3bf + ((size_t)b * NPTS + n) * 64 + 32 * h;
#pragma unroll
  for (int q = 0; q < 4; q++) {
    v8bf nv = *(const v8bf*)(nrow + 8 * q);
    v8bf tv = *(const v8bf*)(trow + 8 * q);
    v8bf dv;
#pragma unroll
    for (int e = 0; e < 8; e++) dv[e] = (bf16)((float)nv[e] - (float)tv[e]);
    *(v8bf*)(&S[nn][0 + 32 * h + 8 * q])   = nv;  // raw neigh (becomes shifted later)
    *(v8bf*)(&S[nn][64 + 32 * h + 8 * q])  = dv;  // neigh - tile
    *(v8bf*)(&S[nn][128 + 32 * h + 8 * q]) = tv;  // tile
  }
  const float* nx = xyz + ((size_t)b * NPTS + idxm) * 3;
  const float* tx = xyz + ((size_t)b * NPTS + n) * 3;
  float nxv[3] = {nx[0], nx[1], nx[2]};
  float txv[3] = {tx[0], tx[1], tx[2]};
  wave_lds_fence();

  // ---- w5: feat_info(ch 64..191) -> 3 xyz offsets ----
  {
    v8f acc = splat8(nn < 3 ? be5[nn] : 0.f);
#pragma unroll
    for (int kc = 0; kc < 4; kc++) {
      v16bf a = load_a_pair(&S[nn][0], 64 + 32 * kc, h);
      v16bf bb = load_b16(W5, 128, nn, 32 * kc, h);
      acc = wmma_bf16(a, bb, acc);
    }
    if (nn < 3) {
#pragma unroll
      for (int v = 0; v < 8; v++) {
        float x = acc[v];
        xoff[wave][v + 8 * h][nn] = x > 0.f ? x : 0.f;
      }
    }
    wave_lds_fence();
  }

  // ---- build xyz_info rows (9 channels + zero pad) ----
  if (h == 0) {
    float ox = xoff[wave][nn][0], oy = xoff[wave][nn][1], oz = xoff[wave][nn][2];
    bf16* r = st2[wave][nn];
    r[0] = (bf16)(nxv[0] - txv[0]); r[1] = (bf16)(nxv[1] - txv[1]); r[2] = (bf16)(nxv[2] - txv[2]);
    r[3] = (bf16)(nxv[0] + ox);     r[4] = (bf16)(nxv[1] + oy);     r[5] = (bf16)(nxv[2] + oz);
    r[6] = (bf16)txv[0];            r[7] = (bf16)txv[1];            r[8] = (bf16)txv[2];
#pragma unroll
    for (int c = 9; c < 32; c++) r[c] = (bf16)0.f;
  }
  wave_lds_fence();

  // ---- w6 (feat offsets -> shifted neigh feat, in-place) and w7 (xyz encoding) ----
  v8f oi8[8];  // overall_info tiles: [0..3]=xyz_encoding, [4..7]=feat_encoding
  {
    v16bf a2 = load_a_pair(&st2[wave][nn][0], 0, h);
#pragma unroll
    for (int t = 0; t < 4; t++) {
      v16bf b6 = load_b16(W6, 32, 16 * t + nn, 0, h);
      v8f acc = splat8(be6[16 * t + nn]);
      acc = wmma_bf16(a2, b6, acc);
#pragma unroll
      for (int v = 0; v < 8; v++) {
        float off = acc[v]; off = off > 0.f ? off : 0.f;
        int m = v + 8 * h, o = 16 * t + nn;
        float nf = (float)S[m][o];
        S[m][o] = (bf16)(nf + off);   // shifted_neigh_feat
      }
      v16bf b7 = load_b16(W7, 32, 16 * t + nn, 0, h);
      v8f acc7 = splat8(be7[16 * t + nn]);
      acc7 = wmma_bf16(a2, b7, acc7);
#pragma unroll
      for (int v = 0; v < 8; v++) { float x = acc7[v]; acc7[v] = x > 0.f ? x : 0.f; }
      oi8[t] = acc7;
    }
  }
  wave_lds_fence();

  // ---- w8: feat_info2 (ch 0..191) -> feat_encoding ----
#pragma unroll
  for (int t = 0; t < 4; t++) oi8[4 + t] = splat8(be8[16 * t + nn]);
#pragma unroll
  for (int k0 = 0; k0 < 192; k0 += 32) {
    v16bf a = load_a_pair(&S[nn][0], k0, h);
#pragma unroll
    for (int t = 0; t < 4; t++) {
      v16bf bb = load_b16(W8, 192, 16 * t + nn, k0, h);
      oi8[4 + t] = wmma_bf16(a, bb, oi8[4 + t]);
    }
  }
#pragma unroll
  for (int t = 0; t < 4; t++)
#pragma unroll
    for (int v = 0; v < 8; v++) { float x = oi8[4 + t][v]; oi8[4 + t][v] = x > 0.f ? x : 0.f; }

  // ---- write overall_info (128ch) into staging for the w9 A operand ----
#pragma unroll
  for (int t = 0; t < 8; t++)
#pragma unroll
    for (int v = 0; v < 8; v++) S[v + 8 * h][16 * t + nn] = (bf16)oi8[t][v];
  wave_lds_fence();

  // ---- w9: attention logits ----
  v8f kw[8];
#pragma unroll
  for (int t = 0; t < 8; t++) kw[t] = splat8(b9[16 * t + nn]);
#pragma unroll
  for (int k0 = 0; k0 < 128; k0 += 32) {
    v16bf a = load_a_pair(&S[nn][0], k0, h);
#pragma unroll
    for (int t = 0; t < 8; t++) {
      v16bf bb = load_b16(W9, 128, 16 * t + nn, k0, h);
      kw[t] = wmma_bf16(a, bb, kw[t]);
    }
  }

  // ---- softmax over 16 neighbors (rows) + weighted sum + max, per channel ----
#pragma unroll
  for (int t = 0; t < 8; t++) {
    v8f oi = oi8[t];
    float mx = -3.4e38f, omax = -3.4e38f;
#pragma unroll
    for (int v = 0; v < 8; v++) { mx = fmaxf(mx, kw[t][v]); omax = fmaxf(omax, oi[v]); }
    mx   = fmaxf(mx,   __shfl_xor(mx, 16, 32));
    omax = fmaxf(omax, __shfl_xor(omax, 16, 32));
    float s = 0.f, ws = 0.f;
#pragma unroll
    for (int v = 0; v < 8; v++) {
      float e = __expf(kw[t][v] - mx);
      s += e; ws += e * oi[v];
    }
    s  += __shfl_xor(s, 16, 32);
    ws += __shfl_xor(ws, 16, 32);
    float wsum = ws / s;
    if (h == 0) {
      erow[wave][16 * t + nn]       = (bf16)omax;
      erow[wave][128 + 16 * t + nn] = (bf16)wsum;
    }
  }
  wave_lds_fence();

  // ---- coalesced 512B store of the 256-ch encoding ----
  *(v8bf*)(encbf + ((size_t)b * NPTS + n) * 256 + 8 * lane) =
      *(const v8bf*)(&erow[wave][8 * lane]);
}

// ---------- kernel 3: per-point MLPs w10 (256->128) and w11 (128->256) ----------
__global__ void k3_out(const bf16* encbf,
                       const bf16* W10, const float* be10,
                       const bf16* W11, const float* be11,
                       float* out) {
  __shared__ bf16 st[WAVES][16][136];
  __shared__ float tr[WAVES][16][17];
  int wave = threadIdx.x >> 5, lane = threadIdx.x & 31;
  int nn = lane & 15, h = lane >> 4;
  int tile = blockIdx.x * WAVES + wave;   // 0..1023
  int b = tile / (NPTS / 16);
  int n0 = (tile % (NPTS / 16)) * 16;

  // warm W10/W11 fragment rows while A rows stream in
#pragma unroll
  for (int t = 0; t < 8; t++) {
    pf(W10 + (size_t)(16 * t + nn) * 256);
    pf(W10 + (size_t)(16 * t + nn) * 256 + 128);
  }
#pragma unroll
  for (int t = 0; t < 8; t++) pf(W11 + (size_t)(32 * t + nn) * 128);

  const bf16* arow = encbf + ((size_t)b * NPTS + n0 + nn) * 256;
  v8f acc[8];
#pragma unroll
  for (int t = 0; t < 8; t++) acc[t] = splat8(be10[16 * t + nn]);
#pragma unroll
  for (int k0 = 0; k0 < 256; k0 += 32) {
    v16bf a = load_a_pair(arow, k0, h);
#pragma unroll
    for (int t = 0; t < 8; t++) {
      v16bf bb = load_b16(W10, 256, 16 * t + nn, k0, h);
      acc[t] = wmma_bf16(a, bb, acc[t]);
    }
  }
#pragma unroll
  for (int t = 0; t < 8; t++)
#pragma unroll
    for (int v = 0; v < 8; v++) {
      float x = acc[t][v];
      st[wave][v + 8 * h][16 * t + nn] = (bf16)(x > 0.f ? x : 0.f);
    }
  wave_lds_fence();

  const bf16* srow = &st[wave][nn][0];
  int oc = lane >> 1, half = lane & 1;
  for (int t2 = 0; t2 < 16; t2++) {
    v8f a2 = splat8(be11[16 * t2 + nn]);
#pragma unroll
    for (int kc = 0; kc < 4; kc++) {
      v16bf a = load_a_pair(srow, 32 * kc, h);
      v16bf bb = load_b16(W11, 128, 16 * t2 + nn, 32 * kc, h);
      a2 = wmma_bf16(a, bb, a2);
    }
#pragma unroll
    for (int v = 0; v < 8; v++) {
      float x = a2[v];
      tr[wave][v + 8 * h][nn] = x > 0.f ? x : 0.f;
    }
    wave_lds_fence();
    float vals[8];
#pragma unroll
    for (int r = 0; r < 8; r++) vals[r] = tr[wave][8 * half + r][oc];
    float* dst = out + ((size_t)b * 256 + 16 * t2 + oc) * NPTS + n0 + 8 * half;
    *(float4*)(dst)     = make_float4(vals[0], vals[1], vals[2], vals[3]);
    *(float4*)(dst + 4) = make_float4(vals[4], vals[5], vals[6], vals[7]);
    wave_lds_fence();
  }
}

// ---------- launch ----------
extern "C" void kernel_launch(void* const* d_in, const int* in_sizes, int n_in,
                              void* d_out, int out_size, void* d_ws, size_t ws_size,
                              hipStream_t stream) {
  (void)in_sizes; (void)n_in; (void)out_size; (void)ws_size;
  const float* feature = (const float*)d_in[0];
  const float* xyz     = (const float*)d_in[1];
  const int*   nidx    = (const int*)d_in[2];
  const float* w1 = (const float*)d_in[3],  *g1 = (const float*)d_in[4],  *be1 = (const float*)d_in[5];
  const float* w5 = (const float*)d_in[6],  *g5 = (const float*)d_in[7],  *be5 = (const float*)d_in[8];
  const float* w6 = (const float*)d_in[9],  *g6 = (const float*)d_in[10], *be6 = (const float*)d_in[11];
  const float* w7 = (const float*)d_in[12], *g7 = (const float*)d_in[13], *be7 = (const float*)d_in[14];
  const float* w8 = (const float*)d_in[15], *g8 = (const float*)d_in[16], *be8 = (const float*)d_in[17];
  const float* w9 = (const float*)d_in[18], *b9 = (const float*)d_in[19];
  const float* w10 = (const float*)d_in[20], *g10 = (const float*)d_in[21], *be10 = (const float*)d_in[22];
  const float* w11 = (const float*)d_in[23], *g11 = (const float*)d_in[24], *be11 = (const float*)d_in[25];

  char* ws = (char*)d_ws;
  size_t off = 0;
  auto take = [&](size_t bytes) -> char* {
    char* p = ws + off;
    off = (off + bytes + 255) & ~(size_t)255;
    return p;
  };
  bf16* f3bf  = (bf16*)take((size_t)NB * NPTS * 64 * 2);
  bf16* encbf = (bf16*)take((size_t)NB * NPTS * 256 * 2);
  bf16* W1  = (bf16*)take(64 * 64 * 2);
  bf16* W5  = (bf16*)take(16 * 128 * 2);
  bf16* W6  = (bf16*)take(64 * 32 * 2);
  bf16* W7  = (bf16*)take(64 * 32 * 2);
  bf16* W8  = (bf16*)take(64 * 192 * 2);
  bf16* W9  = (bf16*)take(128 * 128 * 2);
  bf16* W10 = (bf16*)take(128 * 256 * 2);
  bf16* W11 = (bf16*)take(256 * 128 * 2);

  prep_weights<<<dim3(64), dim3(256), 0, stream>>>(
      w1, g1, w5, g5, w6, g6, w7, g7, w8, g8, w9, w10, g10, w11, g11,
      W1, W5, W6, W7, W8, W9, W10, W11);

  k1_feat<<<dim3((NB * NPTS / 16) / WAVES), dim3(32 * WAVES), 0, stream>>>(
      feature, W1, be1, f3bf);

  k2_point<<<dim3((NB * NPTS) / WAVES), dim3(32 * WAVES), 0, stream>>>(
      f3bf, xyz, nidx, W5, be5, W6, be6, W7, be7, W8, be8, W9, b9, encbf);

  k3_out<<<dim3((NB * NPTS / 16) / WAVES), dim3(32 * WAVES), 0, stream>>>(
      encbf, W10, be10, W11, be11, (float*)d_out);
}
